// MultiHeadAttention_2516850835981
// MI455X (gfx1250) — matrix operand order repaired and checked by reading, not compiled, not run on hardware
//
#include <hip/hip_runtime.h>

typedef __attribute__((ext_vector_type(16))) __bf16 v16bf;
typedef __attribute__((ext_vector_type(8)))  float  v8f;
typedef __attribute__((ext_vector_type(2)))  __bf16 bf16x2;

#define Bc 4
#define Tc 2048
#define Dc 1024
#define Hc 16
#define DKc 64

// ---------------- WMMA helper ----------------
__device__ __forceinline__ v8f wmma_bf16(v16bf a, v16bf b, v8f c) {
  // D = A(16x32 bf16) * B(32x16 bf16) + C(16x16 f32)
  return __builtin_amdgcn_wmma_f32_16x16x32_bf16(
      /*neg_a=*/false, a, /*neg_b=*/false, b,
      /*c_mod=*/(short)0, c, /*reuse_a=*/false, /*reuse_b=*/false);
}

// ---- A fragment (16x32 bf16), contiguous k, not the ISA 7.12.2 table order ----
// Lane l holds row M = l&15. Element i: K = (l>>4)*16 + i.
// K pairs are memory-contiguous -> pairwise loads (compiler merges to b128).
__device__ __forceinline__ v16bf load_fragA(const __bf16* base, int row_stride, int lane) {
  const int half = lane >> 4;
  const __bf16* rowp = base + (size_t)(lane & 15) * row_stride;
  v16bf f;
#pragma unroll
  for (int j = 0; j < 8; ++j) {
    int kb = half * 16 + j * 2;
    bf16x2 v = *(const bf16x2*)(rowp + kb);
    f[2 * j]     = v.x;
    f[2 * j + 1] = v.y;
  }
  return f;
}

// ---- B fragment (32x16 bf16) ----
// this code's order (not AMD's table order): lanes 0-15 hold K=0-15, lanes 16-31 hold K=16-31, sequential.
// Lane l supplies column N = l&15, elements i=0..15 are K=(l>>4)*16+i.
// base points at B^T tile start (row-major, rows = N, cols = K): one contiguous 32B load.
__device__ __forceinline__ v16bf load_fragB(const __bf16* base, int row_stride, int lane) {
  const __bf16* p = base + (size_t)(lane & 15) * row_stride + (lane >> 4) * 16;
  return *(const v16bf*)p;   // 32 bytes, 32B-aligned at every call site
}

// 16-lane half-group reductions (wave32; masks < 16 stay inside the group)
__device__ __forceinline__ float rmax16(float v) {
#pragma unroll
  for (int m = 1; m < 16; m <<= 1) v = fmaxf(v, __shfl_xor(v, m, 32));
  return v;
}
__device__ __forceinline__ float rsum16(float v) {
#pragma unroll
  for (int m = 1; m < 16; m <<= 1) v += __shfl_xor(v, m, 32);
  return v;
}

// ============ Kernel 0: f32 -> bf16 conversion (x, W_qkv, W_proj) ============
__global__ void __launch_bounds__(256) cvt_kernel(const float* __restrict__ in,
                                                  __bf16* __restrict__ out, int n4) {
  const int i = blockIdx.x * 256 + threadIdx.x;
  if (i < n4) {
    float4 v = ((const float4*)in)[i];
    bf16x2 a, b;
    a.x = (__bf16)v.x; a.y = (__bf16)v.y;
    b.x = (__bf16)v.z; b.y = (__bf16)v.w;
    ((bf16x2*)out)[2 * i]     = a;
    ((bf16x2*)out)[2 * i + 1] = b;
  }
}

// ============ Kernel 1: QKV GEMM (Xb[8192,1024] @ Wqkv^T + b) -> Q,K,V^T bf16 ============
// grid (256, 6), block 256 (8 waves). Wave: 32 rows x 64 cols (8 WMMA / k-step, B reused x2).
__global__ void __launch_bounds__(256) qkv_kernel(const __bf16* __restrict__ Xb,
                                                  const __bf16* __restrict__ Wb,
                                                  const float* __restrict__ bqkv,
                                                  __bf16* __restrict__ Q,
                                                  __bf16* __restrict__ K,
                                                  __bf16* __restrict__ Vt) {
  const int lane = threadIdx.x & 31;
  const int wave = threadIdx.x >> 5;
  const int mbase = blockIdx.x * 32;
  const int nbase = blockIdx.y * 512 + wave * 64;
  const __bf16* arow0 = Xb + (size_t)mbase * Dc;
  const __bf16* arow1 = arow0 + (size_t)16 * Dc;

  v8f acc[2][4];
#pragma unroll
  for (int mt = 0; mt < 2; ++mt)
#pragma unroll
    for (int t = 0; t < 4; ++t) acc[mt][t] = (v8f){};

  for (int k0 = 0; k0 < Dc; k0 += 32) {
    v16bf a0 = load_fragA(arow0 + k0, Dc, lane);
    v16bf a1 = load_fragA(arow1 + k0, Dc, lane);
#pragma unroll
    for (int t = 0; t < 4; ++t) {
      v16bf b = load_fragB(Wb + (size_t)(nbase + 16 * t) * Dc + k0, Dc, lane);
      acc[0][t] = wmma_bf16(a0, b, acc[0][t]);
      acc[1][t] = wmma_bf16(a1, b, acc[1][t]);
    }
  }

  const int half = lane >> 4, nl = lane & 15;
#pragma unroll
  for (int t = 0; t < 4; ++t) {
    const int n = nbase + 16 * t + nl;      // 0..3071
    const float bias = bqkv[n];
    const int which = n >> 10;              // 0:Q 1:K 2:V (uniform per fragment)
    const int d = n & (Dc - 1);
    const int h = d >> 6, dk = d & 63;
#pragma unroll
    for (int mt = 0; mt < 2; ++mt)
#pragma unroll
      for (int r = 0; r < 8; ++r) {
        const int row = mbase + mt * 16 + r + half * 8;   // b*T + t
        const int bb = row >> 11, tt = row & (Tc - 1);
        const __bf16 val = (__bf16)(acc[mt][t][r] + bias);
        const size_t bh = (size_t)(bb * Hc + h);
        if (which == 0)      Q [(bh * Tc + tt) * DKc + dk] = val;
        else if (which == 1) K [(bh * Tc + tt) * DKc + dk] = val;
        else                 Vt[(bh * DKc + dk) * Tc + tt] = val;
      }
  }
}

// ============ Kernel 2: causal flash attention ============
// One wave per (b,h, 16-query tile). grid 1024 blocks * 8 waves = 64*128 tiles.
__global__ void __launch_bounds__(256) attn_kernel(const __bf16* __restrict__ Q,
                                                   const __bf16* __restrict__ K,
                                                   const __bf16* __restrict__ Vt,
                                                   __bf16* __restrict__ O) {
  __shared__ __bf16 ldsP[8 * 16 * 32];      // per-wave 16x32 P tile
  const int lane = threadIdx.x & 31;
  const int wave = threadIdx.x >> 5;
  const int tile = blockIdx.x * 8 + wave;
  const int bh = tile >> 7;                 // /(T/16)
  const int qt = tile & 127;
  const int b = bh >> 4, h = bh & (Hc - 1);

  const __bf16* Qb = Q  + ((size_t)bh * Tc + qt * 16) * DKc;
  const __bf16* Kb = K  + (size_t)bh * Tc * DKc;
  const __bf16* Vb = Vt + (size_t)bh * DKc * Tc;
  __bf16* P = ldsP + wave * (16 * 32);

  const v16bf aQ0 = load_fragA(Qb,      DKc, lane);  // dk 0..31
  const v16bf aQ1 = load_fragA(Qb + 32, DKc, lane);  // dk 32..63

  const int half = lane >> 4, nl = lane & 15;
  float m_i[8], l_i[8];
#pragma unroll
  for (int r = 0; r < 8; ++r) { m_i[r] = -1e30f; l_i[r] = 0.f; }
  v8f o0 = {}, o1 = {}, o2 = {}, o3 = {};
  const v8f zero = {};
  const float sc = 0.125f;                  // 1/sqrt(64)

  const int nblk = (qt >> 1) + 1;           // 32-key blocks up to the diagonal
  for (int j = 0; j < nblk; ++j) {
    const int kb0 = j * 32;
    // ---- S = Q @ K^T (two 16x16 tiles); B^T rows are K rows (contiguous dk) ----
    v16bf bK;
    bK = load_fragB(Kb + (size_t)kb0 * DKc,             DKc, lane);
    v8f s0 = wmma_bf16(aQ0, bK, zero);
    bK = load_fragB(Kb + (size_t)kb0 * DKc + 32,        DKc, lane);
    s0 = wmma_bf16(aQ1, bK, s0);
    bK = load_fragB(Kb + (size_t)(kb0 + 16) * DKc,      DKc, lane);
    v8f s1 = wmma_bf16(aQ0, bK, zero);
    bK = load_fragB(Kb + (size_t)(kb0 + 16) * DKc + 32, DKc, lane);
    s1 = wmma_bf16(aQ1, bK, s1);

    // ---- scale + causal mask + online softmax ----
    float mx[8];
#pragma unroll
    for (int r = 0; r < 8; ++r) {
      const int qrow = qt * 16 + r + half * 8;
      float v0 = s0[r] * sc; if (kb0 + nl      > qrow) v0 = -1e30f;
      float v1 = s1[r] * sc; if (kb0 + 16 + nl > qrow) v1 = -1e30f;
      s0[r] = v0; s1[r] = v1;
      mx[r] = fmaxf(v0, v1);
    }
#pragma unroll
    for (int r = 0; r < 8; ++r) {
      const float mnew  = fmaxf(m_i[r], rmax16(mx[r]));
      const float alpha = __expf(m_i[r] - mnew);
      m_i[r] = mnew;
      const float p0 = __expf(s0[r] - mnew);
      const float p1 = __expf(s1[r] - mnew);
      l_i[r] = l_i[r] * alpha + rsum16(p0 + p1);
      o0[r] *= alpha; o1[r] *= alpha; o2[r] *= alpha; o3[r] *= alpha;
      const int m = r + half * 8;           // C-layout row -> LDS row
      P[m * 32 + nl]      = (__bf16)p0;
      P[m * 32 + 16 + nl] = (__bf16)p1;
    }
    asm volatile("s_wait_dscnt 0" ::: "memory");  // LDS RAW: P stores -> A-frag loads

    // ---- O += P(16x32) @ V(32x64); V^T rows (fixed dk) are contiguous in key ----
    const v16bf aP = load_fragA(P, 32, lane);
    v16bf bV;
    bV = load_fragB(Vb + (size_t) 0 * Tc + kb0, Tc, lane); o0 = wmma_bf16(aP, bV, o0);
    bV = load_fragB(Vb + (size_t)16 * Tc + kb0, Tc, lane); o1 = wmma_bf16(aP, bV, o1);
    bV = load_fragB(Vb + (size_t)32 * Tc + kb0, Tc, lane); o2 = wmma_bf16(aP, bV, o2);
    bV = load_fragB(Vb + (size_t)48 * Tc + kb0, Tc, lane); o3 = wmma_bf16(aP, bV, o3);
  }

  // ---- epilogue: normalize, write [B,T,H*dk] bf16 ----
  __bf16* outp = O + ((size_t)(b * Tc + qt * 16)) * Dc + h * DKc;
#pragma unroll
  for (int r = 0; r < 8; ++r) {
    const float inv = 1.0f / l_i[r];
    const size_t rowoff = (size_t)(r + half * 8) * Dc;
    outp[rowoff +  0 + nl] = (__bf16)(o0[r] * inv);
    outp[rowoff + 16 + nl] = (__bf16)(o1[r] * inv);
    outp[rowoff + 32 + nl] = (__bf16)(o2[r] * inv);
    outp[rowoff + 48 + nl] = (__bf16)(o3[r] * inv);
  }
}

// ============ Kernel 3: output projection (attn_out @ Wproj^T + b) -> f32 ============
// grid (256, 2), block 256 (8 waves); wave covers 32 rows x 64 cols.
__global__ void __launch_bounds__(256) proj_kernel(const __bf16* __restrict__ A,
                                                   const __bf16* __restrict__ Wb,
                                                   const float* __restrict__ bp,
                                                   float* __restrict__ out) {
  const int lane = threadIdx.x & 31;
  const int wave = threadIdx.x >> 5;
  const int mbase = blockIdx.x * 32;
  const int nbase = blockIdx.y * 512 + wave * 64;
  const __bf16* arow0 = A + (size_t)mbase * Dc;
  const __bf16* arow1 = arow0 + (size_t)16 * Dc;

  v8f acc[2][4];
#pragma unroll
  for (int mt = 0; mt < 2; ++mt)
#pragma unroll
    for (int t = 0; t < 4; ++t) acc[mt][t] = (v8f){};

  for (int k0 = 0; k0 < Dc; k0 += 32) {
    v16bf a0 = load_fragA(arow0 + k0, Dc, lane);
    v16bf a1 = load_fragA(arow1 + k0, Dc, lane);
#pragma unroll
    for (int t = 0; t < 4; ++t) {
      v16bf b = load_fragB(Wb + (size_t)(nbase + 16 * t) * Dc + k0, Dc, lane);
      acc[0][t] = wmma_bf16(a0, b, acc[0][t]);
      acc[1][t] = wmma_bf16(a1, b, acc[1][t]);
    }
  }

  const int half = lane >> 4, nl = lane & 15;
#pragma unroll
  for (int t = 0; t < 4; ++t) {
    const int n = nbase + 16 * t + nl;
    const float bias = bp[n];
#pragma unroll
    for (int mt = 0; mt < 2; ++mt)
#pragma unroll
      for (int r = 0; r < 8; ++r) {
        const int row = mbase + mt * 16 + r + half * 8;
        out[(size_t)row * Dc + n] = acc[mt][t][r] + bias;
      }
  }
}

extern "C" void kernel_launch(void* const* d_in, const int* in_sizes, int n_in,
                              void* d_out, int out_size, void* d_ws, size_t ws_size,
                              hipStream_t stream) {
  const float* x    = (const float*)d_in[0];
  const float* Wqkv = (const float*)d_in[1];
  const float* bqkv = (const float*)d_in[2];
  const float* Wp   = (const float*)d_in[3];
  const float* bp   = (const float*)d_in[4];
  float* out = (float*)d_out;

  const size_t NEL   = (size_t)Bc * Hc * Tc * DKc;  // 8,388,608
  const size_t NXW   = (size_t)Bc * Tc * Dc;        // 8,388,608 (x elements)
  const size_t NWQKV = (size_t)3 * Dc * Dc;         // 3,145,728
  const size_t NWP   = (size_t)Dc * Dc;             // 1,048,576

  __bf16* Q     = (__bf16*)d_ws;
  __bf16* K     = Q     + NEL;
  __bf16* Vt    = K     + NEL;
  __bf16* AO    = Vt    + NEL;      // [B,T,D] bf16
  __bf16* Xb    = AO    + NEL;
  __bf16* Wqkvb = Xb    + NXW;
  __bf16* Wpb   = Wqkvb + NWQKV;

  cvt_kernel<<<(int)(NXW   / 4 + 255) / 256, 256, 0, stream>>>(x,    Xb,    (int)(NXW   / 4));
  cvt_kernel<<<(int)(NWQKV / 4 + 255) / 256, 256, 0, stream>>>(Wqkv, Wqkvb, (int)(NWQKV / 4));
  cvt_kernel<<<(int)(NWP   / 4 + 255) / 256, 256, 0, stream>>>(Wp,   Wpb,   (int)(NWP   / 4));

  qkv_kernel <<<dim3(256, 6), 256, 0, stream>>>(Xb, Wqkvb, bqkv, Q, K, Vt);
  attn_kernel<<<dim3(1024),   256, 0, stream>>>(Q, K, Vt, AO);
  proj_kernel<<<dim3(256, 2), 256, 0, stream>>>(AO, Wpb, bp, out);
}
